// QuantLinear_16028817949314
// MI455X (gfx1250) — matrix operand order repaired and checked by reading, not compiled, not run on hardware
//
#include <hip/hip_runtime.h>

typedef __attribute__((ext_vector_type(16))) _Float16 v16h;
typedef __attribute__((ext_vector_type(8)))  _Float16 v8h;
typedef __attribute__((ext_vector_type(4)))  _Float16 v4h;
typedef __attribute__((ext_vector_type(8)))  float    v8f;
typedef __attribute__((ext_vector_type(4)))  float    v4f;
typedef __attribute__((ext_vector_type(4)))  int      v4i;

namespace {
constexpr int TOK    = 8192;
constexpr int IND    = 4096;
constexpr int OUTD   = 4096;
constexpr int GROUPS = 32;      // IN / 128

constexpr int BM  = 128;
constexpr int BN  = 128;
constexpr int BK  = 64;         // staged K slab (2 WMMA K-steps of 32)
constexpr int LDT = BK + 8;     // padded LDS row stride in halves (72 -> 144B rows, 16B aligned)
}

__global__ __launch_bounds__(256)
void quantlinear_wmma_f16(const float* __restrict__ x,
                          const int*   __restrict__ qw,
                          const float* __restrict__ scales,
                          const float* __restrict__ bias,
                          float*       __restrict__ out) {
  __shared__ __attribute__((aligned(16))) _Float16 Xs[BM * LDT];
  __shared__ __attribute__((aligned(16))) _Float16 Ws[BN * LDT];

  const int tid   = threadIdx.x;
  const int lane  = tid & 31;
  const int wave  = tid >> 5;
  const int wy    = wave >> 2;   // 0..1  : M direction (64 rows each)
  const int wx    = wave & 3;    // 0..3  : N direction (32 cols each)
  const int lmod  = lane & 15;
  const int lhalf = lane >> 4;   // 0 or 1

  const int blockN = blockIdx.x * BN;
  const int blockM = blockIdx.y * BM;

  v8f acc[4][2];
#pragma unroll
  for (int mi = 0; mi < 4; ++mi)
#pragma unroll
    for (int ni = 0; ni < 2; ++ni)
      acc[mi][ni] = (v8f){0.f, 0.f, 0.f, 0.f, 0.f, 0.f, 0.f, 0.f};

  for (int kk = 0; kk < IND; kk += BK) {
    // BK=64 slab lies entirely inside one 128-wide quant group
    const int gi = kk >> 7;

    // ---- stage tiles into LDS: x f32 -> f16, Q int4-code -> dequant f16 ----
    // 128 rows x 64 cols = 2048 vec4 per tile; 8 vec4 per thread per tile
#pragma unroll
    for (int i = 0; i < 8; ++i) {
      const int idx = i * 256 + tid;   // 0..2047
      const int row = idx >> 4;        // 0..127
      const int c4  = idx & 15;        // 0..15 (vec4 within 64-wide slab)

      // activation tile
      const v4f xv = *(const v4f*)(x + (size_t)(blockM + row) * IND + kk + c4 * 4);
      v4h xh;
#pragma unroll
      for (int e = 0; e < 4; ++e) xh[e] = (_Float16)xv[e];
      *(v4h*)(&Xs[row * LDT + c4 * 4]) = xh;

      // weight tile, dequantized on the fly
      const int   n  = blockN + row;
      const float s  = scales[n * GROUPS + gi];
      const v4i   qv = *(const v4i*)(qw + (size_t)n * IND + kk + c4 * 4);
      v4h wh;
#pragma unroll
      for (int e = 0; e < 4; ++e) wh[e] = (_Float16)((float)qv[e] * s);
      *(v4h*)(&Ws[row * LDT + c4 * 4]) = wh;
    }

    // prefetch next K slab (emits global_prefetch_b8)
    if (kk + BK < IND) {
      __builtin_prefetch((const void*)(x  + (size_t)(blockM + (tid >> 1)) * IND + kk + BK), 0, 0);
      __builtin_prefetch((const void*)(qw + (size_t)(blockN + (tid >> 1)) * IND + kk + BK), 0, 0);
    }
    __syncthreads();

    // ---- two WMMA K-steps of 32 over the staged slab ----
#pragma unroll
    for (int ks = 0; ks < BK; ks += 32) {
      v16h a[4], b[2];
#pragma unroll
      for (int mi = 0; mi < 4; ++mi) {
        // A 16x32 f16: lanes 0-15 -> M=lane, K {0..7,16..23}; lanes 16-31 -> K {8..15,24..31}
        const int row = wy * 64 + mi * 16 + lmod;
        const v8h lo = *(const v8h*)(&Xs[row * LDT + ks + 8 * lhalf]);
        const v8h hi = *(const v8h*)(&Xs[row * LDT + ks + 16 + 8 * lhalf]);
#pragma unroll
        for (int e = 0; e < 8; ++e) { a[mi][e] = lo[e]; a[mi][8 + e] = hi[e]; }
      }
#pragma unroll
      for (int ni = 0; ni < 2; ++ni) {
        // B 32x16 f16: lane holds N=lane%16; lanes 0-15 -> K {0..7,16..23}, lanes 16-31 -> K {8..15,24..31}
        const int n  = wx * 32 + ni * 16 + lmod;
        const v8h lo = *(const v8h*)(&Ws[n * LDT + ks + 8 * lhalf]);
        const v8h hi = *(const v8h*)(&Ws[n * LDT + ks + 16 + 8 * lhalf]);
#pragma unroll
        for (int e = 0; e < 8; ++e) { b[ni][e] = lo[e]; b[ni][8 + e] = hi[e]; }
      }

#pragma unroll
      for (int mi = 0; mi < 4; ++mi)
#pragma unroll
        for (int ni = 0; ni < 2; ++ni)
          acc[mi][ni] = __builtin_amdgcn_wmma_f32_16x16x32_f16(
              false, a[mi], false, b[ni], (short)0, acc[mi][ni], false, false);
    }

    __syncthreads();
  }

  // ---- epilogue: bias add + store (C layout: elem e -> M = e + 8*lhalf, N = lane%16) ----
#pragma unroll
  for (int ni = 0; ni < 2; ++ni) {
    const int   nG = blockN + wx * 32 + ni * 16 + lmod;
    const float bb = bias[nG];
#pragma unroll
    for (int mi = 0; mi < 4; ++mi) {
      const int mBase = blockM + wy * 64 + mi * 16 + 8 * lhalf;
      float* op = out + (size_t)mBase * OUTD + nG;
#pragma unroll
      for (int e = 0; e < 8; ++e)
        op[(size_t)e * OUTD] = acc[mi][ni][e] + bb;
    }
  }
}

extern "C" void kernel_launch(void* const* d_in, const int* in_sizes, int n_in,
                              void* d_out, int out_size, void* d_ws, size_t ws_size,
                              hipStream_t stream) {
  (void)in_sizes; (void)n_in; (void)out_size; (void)d_ws; (void)ws_size;
  const float* x      = (const float*)d_in[0];
  const int*   qw     = (const int*)d_in[1];
  const float* scales = (const float*)d_in[2];
  const float* bias   = (const float*)d_in[3];
  float*       out    = (float*)d_out;

  dim3 grid(OUTD / BN, TOK / BM);   // 32 x 64 = 2048 blocks
  dim3 block(256);                  // 8 wave32 waves
  quantlinear_wmma_f16<<<grid, block, 0, stream>>>(x, qw, scales, bias, out);
}
